// GAT_82910048682363
// MI455X (gfx1250) — compile-verified, matrix operand order checked by reading, not run on hardware
//
#include <hip/hip_runtime.h>
#include <hip/hip_bf16.h>

typedef __attribute__((ext_vector_type(16))) _Float16 v16h;
typedef __attribute__((ext_vector_type(8)))  float    v8f;

#define HEADS   8
#define HID     64
#define DMODEL  512   // HEADS*HID
#define GRAPHS  64
#define CLASSES 10
#define NEG_SLOPE 0.2f

// ---------------- helpers ----------------
__device__ __forceinline__ unsigned f2ord(float f) {
    unsigned u = __float_as_uint(f);
    return (u & 0x80000000u) ? ~u : (u | 0x80000000u);
}
__device__ __forceinline__ float ord2f(unsigned u) {
    return (u & 0x80000000u) ? __uint_as_float(u & 0x7FFFFFFFu) : __uint_as_float(~u);
}

// ---------------- conversion kernels ----------------
__global__ void k_f32_to_f16(const float* __restrict__ in, _Float16* __restrict__ out, long long n) {
    long long t = (long long)blockIdx.x * blockDim.x + threadIdx.x;
    if (t < n) out[t] = (_Float16)in[t];
}

// W: [K, Nn] row-major f32  ->  Wt: [Nn, K] row-major f16
__global__ void k_transpose_f16(const float* __restrict__ W, _Float16* __restrict__ Wt, int K, int Nn) {
    long long t = (long long)blockIdx.x * blockDim.x + threadIdx.x;
    if (t >= (long long)K * Nn) return;
    int k = (int)(t / Nn), n = (int)(t % Nn);
    Wt[(long long)n * K + k] = (_Float16)W[t];
}

// ---------------- WMMA GEMM: C[M,Nn] = A[M,K] @ B[K,Nn] ----------------
// A: f16 row-major.  Bt: f16 = B transposed, [Nn, K] row-major.
// One wave computes a 16x64 strip: 1 A-fragment reused across 4 B-fragments.
__global__ __launch_bounds__(32)
void k_wmma_gemm(const _Float16* __restrict__ A, const _Float16* __restrict__ Bt,
                 float* __restrict__ C, int M, int Nn, int K) {
    const int lane = threadIdx.x & 31;
    const int half = lane >> 4;        // 0 or 1
    const int l16  = lane & 15;
    const int mtile = blockIdx.x;
    if (mtile * 16 >= M) return;
    const int ntile0 = blockIdx.y * 4;
    const int m = mtile * 16 + l16;

    v8f acc[4] = {v8f{}, v8f{}, v8f{}, v8f{}};

    for (int k0 = 0; k0 < K; k0 += 32) {
        // ---- A fragment: lane (half,l16) holds row m, K pairs per ISA layout
        v16h a;
        const _Float16* Arow = A + (long long)m * K + k0;
        #pragma unroll
        for (int v = 0; v < 8; ++v) {
            int kb = ((v & 4) ? 16 : 0) + half * 8 + 2 * (v & 3);
            a[2 * v]     = Arow[kb];
            a[2 * v + 1] = Arow[kb + 1];
        }
        #pragma unroll
        for (int j = 0; j < 4; ++j) {
            int n = (ntile0 + j) * 16 + l16;
            const _Float16* Brow = Bt + (long long)n * K + k0;
            v16h b;
            #pragma unroll
            for (int v = 0; v < 8; ++v) {
                int kb = ((v & 4) ? 16 : 0) + half * 8 + 2 * (v & 3);
                b[2 * v]     = Brow[kb];
                b[2 * v + 1] = Brow[kb + 1];
            }
            acc[j] = __builtin_amdgcn_wmma_f32_16x16x32_f16(
                false, a, false, b, (short)0, acc[j], false, false);
        }
    }
    // ---- store D: lane column l16, VGPR v -> row v + 8*half
    #pragma unroll
    for (int j = 0; j < 4; ++j) {
        int n = (ntile0 + j) * 16 + l16;
        #pragma unroll
        for (int v = 0; v < 8; ++v) {
            int mrow = mtile * 16 + v + 8 * half;
            C[(long long)mrow * Nn + n] = acc[j][v];
        }
    }
}

// ---------------- attention coefficients: a_src/a_dst [N,H] ----------------
__global__ void k_att_coeff(const float* __restrict__ xw,
                            const float* __restrict__ att_src, const float* __restrict__ att_dst,
                            float* __restrict__ asrc, float* __restrict__ adst, int N) {
    int t = blockIdx.x * blockDim.x + threadIdx.x;   // n*HEADS + h
    if (t >= N * HEADS) return;
    int h = t & (HEADS - 1), n = t >> 3;
    const float4* row = (const float4*)(xw + (long long)n * DMODEL + h * HID);
    const float4* as  = (const float4*)(att_src + h * HID);
    const float4* ad  = (const float4*)(att_dst + h * HID);
    float s = 0.f, d = 0.f;
    #pragma unroll
    for (int i = 0; i < HID / 4; ++i) {
        float4 v = row[i], a = as[i], b = ad[i];
        s += v.x * a.x + v.y * a.y + v.z * a.z + v.w * a.w;
        d += v.x * b.x + v.y * b.y + v.z * b.z + v.w * b.w;
    }
    asrc[t] = s;
    adst[t] = d;
}

// ---------------- edge pass 1: segment max (ordered-uint atomicMax) ----------------
__global__ void k_edge_max(const int* __restrict__ ei, int E, int N,
                           const float* __restrict__ asrc, const float* __restrict__ adst,
                           unsigned* __restrict__ mkey) {
    int t = blockIdx.x * blockDim.x + threadIdx.x;   // e*HEADS + h
    int ET = E + N;
    if (t >= ET * HEADS) return;
    int h = t & (HEADS - 1), e = t >> 3;
    int s, d;
    if (e < E) { s = ei[e]; d = ei[E + e]; } else { s = d = e - E; }
    float v = asrc[s * HEADS + h] + adst[d * HEADS + h];
    v = v > 0.f ? v : NEG_SLOPE * v;
    atomicMax(mkey + d * HEADS + h, f2ord(v));
}

// ---------------- edge pass 2: exp + segment sum ----------------
__global__ void k_edge_exp(const int* __restrict__ ei, int E, int N,
                           const float* __restrict__ asrc, const float* __restrict__ adst,
                           const unsigned* __restrict__ mkey,
                           float* __restrict__ exbuf, float* __restrict__ ssum) {
    int t = blockIdx.x * blockDim.x + threadIdx.x;
    int ET = E + N;
    if (t >= ET * HEADS) return;
    int h = t & (HEADS - 1), e = t >> 3;
    int s, d;
    if (e < E) { s = ei[e]; d = ei[E + e]; } else { s = d = e - E; }
    float v = asrc[s * HEADS + h] + adst[d * HEADS + h];
    v = v > 0.f ? v : NEG_SLOPE * v;
    float m  = ord2f(mkey[d * HEADS + h]);
    float ex = __expf(v - m);
    exbuf[t] = ex;
    atomicAdd(ssum + d * HEADS + h, ex);
}

// ---------------- edge pass 3: weighted aggregation (heavy, L2-resident) ----------------
__global__ __launch_bounds__(256)
void k_edge_aggregate(const int* __restrict__ ei, int E, int N,
                      const float* __restrict__ xw, const float* __restrict__ exbuf,
                      const float* __restrict__ ssum, float* __restrict__ agg) {
    // 128 threads per edge (4 channels each), 2 edges per block
    int local = threadIdx.x & 127;
    int e = blockIdx.x * 2 + (threadIdx.x >> 7);
    int ET = E + N;
    if (e >= ET) return;
    int s, d;
    if (e < E) { s = ei[e]; d = ei[E + e]; } else { s = d = e - E; }
    int idx = local * 4;               // 0..508, 4 | HID so one head per thread
    int h = idx >> 6;
    float alpha = exbuf[e * HEADS + h] / ssum[d * HEADS + h];
    const float4 v = *(const float4*)(xw + (long long)s * DMODEL + idx);
    float* o = agg + (long long)d * DMODEL + idx;
    atomicAdd(o + 0, v.x * alpha);
    atomicAdd(o + 1, v.y * alpha);
    atomicAdd(o + 2, v.z * alpha);
    atomicAdd(o + 3, v.w * alpha);
}

// ---------------- bias + ELU (+ optional f32 / f16 outputs) ----------------
__global__ void k_bias_elu(const float* __restrict__ agg, const float* __restrict__ bias,
                           float* __restrict__ outf, _Float16* __restrict__ outh, long long total) {
    long long t = (long long)blockIdx.x * blockDim.x + threadIdx.x;
    if (t >= total) return;
    float v = agg[t] + bias[t & (DMODEL - 1)];
    v = v > 0.f ? v : (__expf(v) - 1.0f);     // ELU alpha=1
    if (outf) outf[t] = v;
    if (outh) outh[t] = (_Float16)v;
}

// ---------------- global mean pool ----------------
__global__ void k_pool_accum(const float* __restrict__ h, const int* __restrict__ batch,
                             float* __restrict__ pooled, float* __restrict__ cnt, int N) {
    int t = blockIdx.x * blockDim.x + threadIdx.x;   // n*(DMODEL/4) + c4
    if (t >= N * (DMODEL / 4)) return;
    int n = t >> 7, c = (t & 127) * 4;
    int g = batch[n];
    const float4 v = *(const float4*)(h + (long long)n * DMODEL + c);
    float* p = pooled + g * DMODEL + c;
    atomicAdd(p + 0, v.x); atomicAdd(p + 1, v.y);
    atomicAdd(p + 2, v.z); atomicAdd(p + 3, v.w);
    if (c == 0) atomicAdd(cnt + g, 1.0f);
}

// ---------------- classifier head + log_softmax ----------------
__global__ __launch_bounds__(32)
void k_head(const float* __restrict__ pooled, const float* __restrict__ cnt,
            const float* __restrict__ lin_w, const float* __restrict__ lin_b,
            float* __restrict__ out) {
    int g = blockIdx.x;
    __shared__ float logits[16];
    int c = threadIdx.x;
    float inv = 1.0f / fmaxf(cnt[g], 1.0f);
    if (c < CLASSES) {
        float acc = lin_b[c];
        for (int dd = 0; dd < DMODEL; ++dd)
            acc += pooled[g * DMODEL + dd] * inv * lin_w[dd * CLASSES + c];
        logits[c] = acc;
    }
    __syncthreads();
    if (c < CLASSES) {
        float m = -3.4e38f;
        for (int i = 0; i < CLASSES; ++i) m = fmaxf(m, logits[i]);
        float s = 0.f;
        for (int i = 0; i < CLASSES; ++i) s += __expf(logits[i] - m);
        out[g * CLASSES + c] = logits[c] - m - __logf(s);
    }
}

// ---------------- orchestration ----------------
static inline int ceil_div(long long a, int b) { return (int)((a + b - 1) / b); }

extern "C" void kernel_launch(void* const* d_in, const int* in_sizes, int n_in,
                              void* d_out, int out_size, void* d_ws, size_t ws_size,
                              hipStream_t stream) {
    const float* x        = (const float*)d_in[0];
    const int*   ei       = (const int*)  d_in[1];
    const int*   batch    = (const int*)  d_in[2];
    const float* W1       = (const float*)d_in[3];
    const float* att_src1 = (const float*)d_in[4];
    const float* att_dst1 = (const float*)d_in[5];
    const float* bias1    = (const float*)d_in[6];
    const float* W2       = (const float*)d_in[7];
    const float* att_src2 = (const float*)d_in[8];
    const float* att_dst2 = (const float*)d_in[9];
    const float* bias2    = (const float*)d_in[10];
    const float* lin_w    = (const float*)d_in[11];
    const float* lin_b    = (const float*)d_in[12];

    const int IN_DIM = 256;
    const int N  = in_sizes[0] / IN_DIM;   // 20000
    const int E  = in_sizes[1] / 2;        // 320000
    const int ET = E + N;                  // + self loops

    // ---- carve workspace (256B aligned) ----
    char* p = (char*)d_ws;
    auto carve = [&](size_t bytes) -> void* {
        void* r = (void*)p;
        p += (bytes + 255) & ~(size_t)255;
        return r;
    };
    _Float16* x16   = (_Float16*)carve((size_t)N * IN_DIM * 2);
    _Float16* w1t   = (_Float16*)carve((size_t)DMODEL * IN_DIM * 2);
    _Float16* w2t   = (_Float16*)carve((size_t)DMODEL * DMODEL * 2);
    _Float16* h16   = (_Float16*)carve((size_t)N * DMODEL * 2);
    float*    xw    = (float*)   carve((size_t)N * DMODEL * 4);
    float*    agg   = (float*)   carve((size_t)N * DMODEL * 4);
    float*    h2    = (float*)   carve((size_t)N * DMODEL * 4);
    float*    asrc  = (float*)   carve((size_t)N * HEADS * 4);
    float*    adst  = (float*)   carve((size_t)N * HEADS * 4);
    unsigned* mkey  = (unsigned*)carve((size_t)N * HEADS * 4);
    float*    ssum  = (float*)   carve((size_t)N * HEADS * 4);
    float*    exbuf = (float*)   carve((size_t)ET * HEADS * 4);
    float*    pooled= (float*)   carve((size_t)GRAPHS * DMODEL * 4);
    float*    cnt   = (float*)   carve((size_t)GRAPHS * 4);

    const int B = 256;
    const dim3 gemm_grid(N / 16, DMODEL / 64);

    // ---- f16 staging: x, W1^T, W2^T ----
    k_f32_to_f16<<<ceil_div((long long)N * IN_DIM, B), B, 0, stream>>>(x, x16, (long long)N * IN_DIM);
    k_transpose_f16<<<ceil_div((long long)IN_DIM * DMODEL, B), B, 0, stream>>>(W1, w1t, IN_DIM, DMODEL);
    k_transpose_f16<<<ceil_div((long long)DMODEL * DMODEL, B), B, 0, stream>>>(W2, w2t, DMODEL, DMODEL);

    // ================= layer 1 =================
    k_wmma_gemm<<<gemm_grid, 32, 0, stream>>>(x16, w1t, xw, N, DMODEL, IN_DIM);
    k_att_coeff<<<ceil_div((long long)N * HEADS, B), B, 0, stream>>>(xw, att_src1, att_dst1, asrc, adst, N);
    hipMemsetAsync(mkey, 0, (size_t)N * HEADS * 4, stream);
    hipMemsetAsync(ssum, 0, (size_t)N * HEADS * 4, stream);
    hipMemsetAsync(agg,  0, (size_t)N * DMODEL * 4, stream);
    k_edge_max<<<ceil_div((long long)ET * HEADS, B), B, 0, stream>>>(ei, E, N, asrc, adst, mkey);
    k_edge_exp<<<ceil_div((long long)ET * HEADS, B), B, 0, stream>>>(ei, E, N, asrc, adst, mkey, exbuf, ssum);
    k_edge_aggregate<<<(ET + 1) / 2, 256, 0, stream>>>(ei, E, N, xw, exbuf, ssum, agg);
    k_bias_elu<<<ceil_div((long long)N * DMODEL, B), B, 0, stream>>>(agg, bias1, nullptr, h16, (long long)N * DMODEL);

    // ================= layer 2 =================
    k_wmma_gemm<<<gemm_grid, 32, 0, stream>>>(h16, w2t, xw, N, DMODEL, DMODEL);
    k_att_coeff<<<ceil_div((long long)N * HEADS, B), B, 0, stream>>>(xw, att_src2, att_dst2, asrc, adst, N);
    hipMemsetAsync(mkey, 0, (size_t)N * HEADS * 4, stream);
    hipMemsetAsync(ssum, 0, (size_t)N * HEADS * 4, stream);
    hipMemsetAsync(agg,  0, (size_t)N * DMODEL * 4, stream);
    k_edge_max<<<ceil_div((long long)ET * HEADS, B), B, 0, stream>>>(ei, E, N, asrc, adst, mkey);
    k_edge_exp<<<ceil_div((long long)ET * HEADS, B), B, 0, stream>>>(ei, E, N, asrc, adst, mkey, exbuf, ssum);
    k_edge_aggregate<<<(ET + 1) / 2, 256, 0, stream>>>(ei, E, N, xw, exbuf, ssum, agg);
    k_bias_elu<<<ceil_div((long long)N * DMODEL, B), B, 0, stream>>>(agg, bias2, h2, nullptr, (long long)N * DMODEL);

    // ================= pool + head =================
    hipMemsetAsync(pooled, 0, (size_t)GRAPHS * DMODEL * 4, stream);
    hipMemsetAsync(cnt,    0, (size_t)GRAPHS * 4, stream);
    k_pool_accum<<<ceil_div((long long)N * (DMODEL / 4), B), B, 0, stream>>>(h2, batch, pooled, cnt, N);
    k_head<<<GRAPHS, 32, 0, stream>>>(pooled, cnt, lin_w, lin_b, (float*)d_out);
}
